// GraphSAGEReasoner_27960237097614
// MI455X (gfx1250) — matrix operand order repaired and checked by reading, not compiled, not run on hardware
//
#include <hip/hip_runtime.h>
#include <hip/hip_bf16.h>

// ---------------- problem constants ----------------
#define NB    4096          // B paths
#define NS    3             // LSTM steps
#define NK    32            // sampled neighbors
#define EMB   128
#define WID   256           // 2*EMB
#define BS    (NB * NS)     // 12288 aggregated rows

typedef __attribute__((ext_vector_type(16))) __bf16 v16bf;
typedef __attribute__((ext_vector_type(8)))  __bf16 v8bf;
typedef __attribute__((ext_vector_type(8)))  float  v8f;

__device__ __forceinline__ __bf16 f2bf(float f) {
    unsigned u = __builtin_bit_cast(unsigned, f);
    unsigned r = u + 0x7FFFu + ((u >> 16) & 1u);     // round-to-nearest-even
    unsigned short h = (unsigned short)(r >> 16);
    return __builtin_bit_cast(__bf16, h);
}

__device__ __forceinline__ float sigmoidf_(float x) {
    return 1.0f / (1.0f + expf(-x));
}

// ---------------- zero init (u32 granularity) ----------------
__global__ void zero_u32(unsigned* __restrict__ p, int n) {
    int i = blockIdx.x * blockDim.x + threadIdx.x;
    if (i < n) p[i] = 0u;
}

// ---------------- weight pack: f32 [K,N] -> per-lane WMMA B fragments (bf16) ----
// Fragment for (tile tn, chunk kc, lane): 16 contiguous bf16 = B[k][col],
// k = kc*32 + 16*(lane>>4) + idx, col = tn*16 + (lane&15).  Zero-pad k >= K.
__global__ __launch_bounds__(256) void pack_weight(
    const float* __restrict__ W, __bf16* __restrict__ Bp,
    int K, int N, int nch_total, int kc_base, int nch_here)
{
    int id   = blockIdx.x * blockDim.x + threadIdx.x;
    int lane = id & 31;
    int rest = id >> 5;
    int kc   = rest % nch_here;
    int tn   = rest / nch_here;
    int tilesN = N >> 4;
    if (tn >= tilesN) return;
    int col = tn * 16 + (lane & 15);
    int kb  = kc * 32 + 16 * (lane >> 4);
    __bf16* dst = Bp + ((size_t)(tn * nch_total + kc_base + kc) * 32 + lane) * 16;
    #pragma unroll
    for (int i = 0; i < 16; ++i) {
        int k = kb + i;
        dst[i] = f2bf((k < K) ? W[(size_t)k * N + col] : 0.0f);
    }
}

// ---------------- gather + vertical mean -> bf16 ----------------
__global__ __launch_bounds__(256) void gather_mean(
    const int* __restrict__ nbr_ent, const int* __restrict__ nbr_rel,
    const float* __restrict__ ent_table, const float* __restrict__ rel_table,
    __bf16* __restrict__ aggb)
{
    __shared__ int se[NK], sr[NK];
    const int bs = blockIdx.x;
    const int t  = threadIdx.x;
    if (t < NK)          se[t]      = nbr_ent[bs * NK + t];
    else if (t < 2 * NK) sr[t - NK] = nbr_rel[bs * NK + (t - NK)];
    __syncthreads();

    float sum = 0.0f;
    if (t < EMB) {
        #pragma unroll 4
        for (int k = 0; k < NK; ++k)
            sum += ent_table[(size_t)se[k] * EMB + t];
    } else {
        const int f = t - EMB;
        #pragma unroll 4
        for (int k = 0; k < NK; ++k)
            sum += rel_table[(size_t)sr[k] * EMB + f];
    }
    aggb[(size_t)bs * WID + t] = f2bf(sum * (1.0f / (float)NK));
}

// ---------------- bf16 WMMA GEMM, all-compile-time shape ----------------
// C = act( concatK(A0, A1) @ Bp + bias ).  A row-major bf16; Bp pre-packed
// fragments.  One wave32 per 16x16 tile; KCH fully-unrolled 32-wide chunks.
template<int N, int KCH, int KSPLIT_CH, int LDA0, int LDA1, int LDC, int ACT, int OBF16>
__global__ __launch_bounds__(256) void gemm_wmma(
    const __bf16* __restrict__ A0, const __bf16* __restrict__ A1,
    const __bf16* __restrict__ Bp, const float* __restrict__ bias,
    float* __restrict__ Cf, __bf16* __restrict__ Cb, int M)
{
    const int lane   = threadIdx.x & 31;
    const int waveId = (int)((blockIdx.x * blockDim.x + threadIdx.x) >> 5);
    const int tilesN = N >> 4;
    const int nTiles = (M >> 4) * tilesN;
    if (waveId >= nTiles) return;          // wave-uniform -> EXEC full at WMMA

    const int tm   = waveId / tilesN;
    const int tn   = waveId - tm * tilesN;
    const int half = lane >> 4;
    const int mr   = lane & 15;
    const int col  = tn * 16 + mr;

    v8f acc = {0.f, 0.f, 0.f, 0.f, 0.f, 0.f, 0.f, 0.f};

    const __bf16* brow0 = A0 + (size_t)(tm * 16 + mr) * LDA0 + 16 * half;
    const __bf16* brow1 = A1 ? (A1 + (size_t)(tm * 16 + mr) * LDA1 + 16 * half) : nullptr;
    const __bf16* bfrag = Bp + ((size_t)tn * KCH * 32 + lane) * 16;

    #pragma unroll
    for (int ch = 0; ch < KCH; ++ch) {
        const __bf16* ar = (ch < KSPLIT_CH) ? brow0 : brow1;
        const int klocal = ((ch < KSPLIT_CH) ? ch : (ch - KSPLIT_CH)) * 32;

        // A fragment: two contiguous 8-element (16B) runs per lane
        v8bf lo = *(const v8bf*)(ar + klocal);
        v8bf hi = *(const v8bf*)(ar + klocal + 8);
        v16bf a;
        #pragma unroll
        for (int i = 0; i < 8; ++i) { a[i] = lo[i]; a[8 + i] = hi[i]; }

        // B fragment: one contiguous 16-element (32B) run per lane
        v16bf b = *(const v16bf*)(bfrag + (size_t)ch * 32 * 16);

        acc = __builtin_amdgcn_wmma_f32_16x16x32_bf16(
                  false, a, false, b, (short)0, acc, false, false);
    }

    const float bv = bias[col];
    #pragma unroll
    for (int r = 0; r < 8; ++r) {
        int   row = tm * 16 + r + 8 * half;
        float v   = acc[r] + bv;
        if (ACT) v = fmaxf(v, 0.0f);
        if (OBF16) Cb[(size_t)row * LDC + col] = f2bf(v);
        else       Cf[(size_t)row * LDC + col] = v;
    }
}

// ---------------- LSTM gates (keras order i,f,g,o); h out in bf16 ----------
__global__ __launch_bounds__(256) void lstm_gates(
    const float* __restrict__ z, float* __restrict__ c, __bf16* __restrict__ h)
{
    int idx = blockIdx.x * blockDim.x + threadIdx.x;     // [0, NB*WID)
    int row = idx >> 8;          // WID == 256
    int j   = idx & 255;
    const float* zr = z + (size_t)row * (4 * WID);
    float gi = zr[j];
    float gf = zr[WID + j];
    float gg = zr[2 * WID + j];
    float go = zr[3 * WID + j];
    float cn = sigmoidf_(gf) * c[idx] + sigmoidf_(gi) * tanhf(gg);
    c[idx] = cn;
    h[idx] = f2bf(sigmoidf_(go) * tanhf(cn));
}

// ---------------- final projection + softmax (N=2) ----------------
__global__ __launch_bounds__(256) void head_softmax(
    const float* __restrict__ x, const float* __restrict__ W3,
    const float* __restrict__ b3, float* __restrict__ out)
{
    int wave = (int)((blockIdx.x * blockDim.x + threadIdx.x) >> 5);
    int lane = threadIdx.x & 31;
    if (wave >= NB) return;
    const float* xr = x + (size_t)wave * 400;
    float s0 = 0.f, s1 = 0.f;
    for (int k = lane; k < 400; k += 32) {
        float v = xr[k];
        s0 += v * W3[k * 2 + 0];
        s1 += v * W3[k * 2 + 1];
    }
    #pragma unroll
    for (int off = 16; off > 0; off >>= 1) {
        s0 += __shfl_down(s0, off);
        s1 += __shfl_down(s1, off);
    }
    if (lane == 0) {
        float l0 = s0 + b3[0], l1 = s1 + b3[1];
        float m  = fmaxf(l0, l1);
        float e0 = expf(l0 - m), e1 = expf(l1 - m);
        float inv = 1.0f / (e0 + e1);
        out[wave * 2 + 0] = e0 * inv;
        out[wave * 2 + 1] = e1 * inv;
    }
}

// ---------------- launch ----------------
extern "C" void kernel_launch(void* const* d_in, const int* in_sizes, int n_in,
                              void* d_out, int out_size, void* d_ws, size_t ws_size,
                              hipStream_t stream) {
    (void)in_sizes; (void)n_in; (void)out_size; (void)ws_size;

    const int*   nbr_ent   = (const int*)  d_in[0];
    const int*   nbr_rel   = (const int*)  d_in[1];
    const float* ent_table = (const float*)d_in[2];
    const float* rel_table = (const float*)d_in[3];
    const float* W_agg     = (const float*)d_in[4];
    const float* b_agg     = (const float*)d_in[5];
    const float* Wx        = (const float*)d_in[6];
    const float* Wh        = (const float*)d_in[7];
    const float* b_lstm    = (const float*)d_in[8];
    const float* W1        = (const float*)d_in[9];
    const float* b1        = (const float*)d_in[10];
    const float* W2        = (const float*)d_in[11];
    const float* b2        = (const float*)d_in[12];
    const float* W3        = (const float*)d_in[13];
    const float* b3        = (const float*)d_in[14];
    float*       out       = (float*)d_out;

    // -------- workspace layout (bytes; every region 256B-multiple) --------
    char* p = (char*)d_ws;
    __bf16* aggb   = (__bf16*)p;  p += (size_t)BS * WID * 2;        // [12288,256] bf16
    __bf16* stepxb = (__bf16*)p;  p += (size_t)BS * WID * 2;        // [B,S,256]  bf16
    float*  z      = (float*) p;  p += (size_t)NB * 4 * WID * 4;    // [B,1024]   f32
    // --- contiguous zero-init region: h(bf16), c(f32), x1(bf16 padded) ---
    char*   zbase  = p;
    __bf16* hb     = (__bf16*)p;  p += (size_t)NB * WID * 2;        // [B,256]  bf16
    float*  c      = (float*) p;  p += (size_t)NB * WID * 4;        // [B,256]  f32
    __bf16* x1b    = (__bf16*)p;  p += (size_t)NB * 416 * 2;        // [B,416]  bf16 (pad)
    size_t  zbytes = (size_t)(p - zbase);
    float*  x2     = (float*) p;  p += (size_t)NB * 400 * 4;        // [B,400]  f32
    __bf16* WaggP  = (__bf16*)p;  p += (size_t)16 * 8  * 32 * 16 * 2;  // 256x256
    __bf16* WxhP   = (__bf16*)p;  p += (size_t)64 * 16 * 32 * 16 * 2;  // 512x1024
    __bf16* W1P    = (__bf16*)p;  p += (size_t)25 * 8  * 32 * 16 * 2;  // 256x400
    __bf16* W2P    = (__bf16*)p;  p += (size_t)25 * 13 * 32 * 16 * 2;  // 416x400

    auto gemm_blocks = [](int M, int N) { return ((M / 16) * (N / 16) + 7) / 8; };
    auto pack_blocks = [](int tilesN, int nch) { return (tilesN * nch * 32 + 255) / 256; };

    // zero h, c, x1(padding) in one pass
    {
        int n = (int)(zbytes / 4);
        zero_u32<<<(n + 255) / 256, 256, 0, stream>>>((unsigned*)zbase, n);
    }

    // pack weights into WMMA B-fragment layout (bf16, zero-padded)
    pack_weight<<<pack_blocks(16, 8),  256, 0, stream>>>(W_agg, WaggP, 256, 256,  8, 0, 8);
    pack_weight<<<pack_blocks(64, 8),  256, 0, stream>>>(Wx,    WxhP,  256, 1024, 16, 0, 8);
    pack_weight<<<pack_blocks(64, 8),  256, 0, stream>>>(Wh,    WxhP,  256, 1024, 16, 8, 8);
    pack_weight<<<pack_blocks(25, 8),  256, 0, stream>>>(W1,    W1P,   256, 400,  8, 0, 8);
    pack_weight<<<pack_blocks(25, 13), 256, 0, stream>>>(W2,    W2P,   400, 400,  13, 0, 13);

    // GraphSAGE gather + mean -> agg (bf16)
    gather_mean<<<BS, 256, 0, stream>>>(nbr_ent, nbr_rel, ent_table, rel_table, aggb);

    // step_x = relu(agg @ W_agg + b_agg)  [12288,256]x[256,256] -> bf16
    gemm_wmma<256, 8, 8, 256, 0, 256, 1, 1><<<gemm_blocks(BS, 256), 256, 0, stream>>>(
        aggb, nullptr, WaggP, b_agg, nullptr, stepxb, BS);

    // LSTM: z = x_s @ Wx + h @ Wh + b   (split-K, compile-time chunk switch)
    for (int s = 0; s < NS; ++s) {
        gemm_wmma<1024, 16, 8, NS * WID, WID, 1024, 0, 0>
            <<<gemm_blocks(NB, 1024), 256, 0, stream>>>(
                stepxb + (size_t)s * WID, hb, WxhP, b_lstm, z, nullptr, NB);
        lstm_gates<<<(NB * WID) / 256, 256, 0, stream>>>(z, c, hb);
    }

    // classifier
    gemm_wmma<400, 8, 8, 256, 0, 416, 1, 1><<<gemm_blocks(NB, 400), 256, 0, stream>>>(
        hb, nullptr, W1P, b1, nullptr, x1b, NB);
    gemm_wmma<400, 13, 13, 416, 0, 400, 1, 0><<<gemm_blocks(NB, 400), 256, 0, stream>>>(
        x1b, nullptr, W2P, b2, x2, nullptr, NB);

    head_softmax<<<(NB * 32 + 255) / 256, 256, 0, stream>>>(x2, W3, b3, out);
}
